// Indexer_50113678409881
// MI455X (gfx1250) — matrix operand order repaired and checked by reading, not compiled, not run on hardware
//
#include <hip/hip_runtime.h>

// ---------------- problem constants ----------------
constexpr int NB   = 2;
constexpr int NS   = 1024;
constexpr int ND   = 7168;
constexpr int NH   = 64;
constexpr int NHD  = 128;
constexpr int NQLR = 1536;
constexpr int NTOPK = 512;
constexpr int NHHD = NH * NHD;     // 8192
constexpr float kNegInf = -1e30f;
constexpr float kEps = 1e-6f;
constexpr float kWScale = 0.011048543456039806f;  // (H*HD)^-0.5

typedef __attribute__((ext_vector_type(16))) __bf16 v16bf;
typedef __attribute__((ext_vector_type(8)))  __bf16 v8bf;
typedef __attribute__((ext_vector_type(8)))  float  v8f;

// CDNA5 async global->LDS path (gfx1250), guarded so a missing builtin
// falls back to a plain VGPR-bounce copy.
#if defined(__HIP_DEVICE_COMPILE__) && defined(__gfx1250__) &&            \
    __has_builtin(__builtin_amdgcn_global_load_async_to_lds_b128) &&       \
    __has_builtin(__builtin_amdgcn_s_wait_asynccnt)
#define USE_ASYNC_LDS 1
typedef int v4i __attribute__((vector_size(16)));
typedef __attribute__((address_space(1))) v4i* as1_b128_t;  // global b128
typedef __attribute__((address_space(3))) v4i* as3_b128_t;  // LDS b128
#else
#define USE_ASYNC_LDS 0
#endif

// ============================================================
// Kernel 0: fp32 -> bf16 bulk convert (for qr and wq operands)
// ============================================================
__global__ void __launch_bounds__(256) f32_to_bf16(
    const float* __restrict__ src, __bf16* __restrict__ dst) {
  int i = blockIdx.x * 256 + threadIdx.x;  // grid sized exactly
  dst[i] = (__bf16)src[i];
}

// ============================================================
// Kernel 1: k = LN(x @ Wk^T) with RoPE  ->  bf16
//           w = (x @ Wproj^T) * (H*HD)^-0.5
// One block per token; x-row staged in LDS (28 KB).
// ============================================================
__global__ void __launch_bounds__(256) prep_kw(
    const float* __restrict__ x, const float* __restrict__ wk,
    const float* __restrict__ lnw, const float* __restrict__ lnb,
    const float* __restrict__ wproj, const float* __restrict__ fcos,
    const float* __restrict__ fsin, __bf16* __restrict__ kout,
    float* __restrict__ wout) {
  __shared__ float xs[ND];
  __shared__ float res[NHD + NH];   // 128 k-values + 64 weights
  __shared__ float kv[NHD];
  __shared__ float stats[2];
  const int tid = threadIdx.x;
  const int tok = blockIdx.x;                 // b*S + t
  const int t = tok & (NS - 1);
  const float* xrow = x + (size_t)tok * ND;
  for (int i = tid; i < ND; i += 256) xs[i] = xrow[i];
  __syncthreads();

  const int lane = tid & 31, wave = tid >> 5;
  for (int o = wave; o < NHD + NH; o += 8) {
    const float* wrow = (o < NHD) ? (wk + (size_t)o * ND)
                                  : (wproj + (size_t)(o - NHD) * ND);
    float acc = 0.f;
    for (int d = lane; d < ND; d += 32) acc += xs[d] * wrow[d];
    for (int off = 16; off > 0; off >>= 1) acc += __shfl_down(acc, off);
    if (lane == 0) res[o] = acc;
  }
  __syncthreads();

  if (tid < 32) {  // mean / var over 128 k-values
    float v0 = res[tid], v1 = res[tid + 32], v2 = res[tid + 64], v3 = res[tid + 96];
    float s1 = v0 + v1 + v2 + v3;
    float s2 = v0 * v0 + v1 * v1 + v2 * v2 + v3 * v3;
    for (int off = 16; off > 0; off >>= 1) {
      s1 += __shfl_down(s1, off);
      s2 += __shfl_down(s2, off);
    }
    if (tid == 0) {
      float mu = s1 * (1.f / NHD);
      stats[0] = mu;
      stats[1] = s2 * (1.f / NHD) - mu * mu;
    }
  }
  __syncthreads();
  if (tid < NHD) {
    float r = rsqrtf(stats[1] + kEps);
    kv[tid] = (res[tid] - stats[0]) * r * lnw[tid] + lnb[tid];
  }
  __syncthreads();
  if (tid < NHD) {  // RoPE on first 64 dims (halves of 32)
    float v;
    if (tid < 32) {
      float c = fcos[t * 32 + tid], sn = fsin[t * 32 + tid];
      v = kv[tid] * c - kv[tid + 32] * sn;
    } else if (tid < 64) {
      int j = tid - 32;
      float c = fcos[t * 32 + j], sn = fsin[t * 32 + j];
      v = kv[j] * sn + kv[tid] * c;
    } else {
      v = kv[tid];
    }
    kout[(size_t)tok * NHD + tid] = (__bf16)v;
  }
  if (tid < NH) wout[(size_t)tok * NH + tid] = res[NHD + tid] * kWScale;
}

// ============================================================
// Kernel 2: q_raw[bs, o] = qr[bs,:] @ wq[o,:]   (2048 x 8192 x 1536)
// bf16 operands (pre-converted). Each wave: 16x64 tile (4 accumulators),
// A loaded once per K-step and reused over 4 v_wmma_f32_16x16x32_bf16.
// ============================================================
__global__ void __launch_bounds__(256) qgemm(
    const __bf16* __restrict__ qrb, const __bf16* __restrict__ wqb,
    float* __restrict__ qraw) {
  const int lane = threadIdx.x & 31;
  const int wave = threadIdx.x >> 5;
  const int half = lane >> 4;          // 0: lanes 0-15, 1: lanes 16-31
  const int l16  = lane & 15;
  const int n_base = (blockIdx.x * 8 + wave) * 64;  // 4 consecutive N tiles
  const int m_base = blockIdx.y * 16;               // token base (b*S+s)
  const __bf16* arow = qrb + (size_t)(m_base + l16) * NQLR;
  const __bf16* brow[4];
#pragma unroll
  for (int j = 0; j < 4; ++j)
    brow[j] = wqb + (size_t)(n_base + j * 16 + l16) * NQLR + half * 16;

  v8f acc0 = {0.f, 0.f, 0.f, 0.f, 0.f, 0.f, 0.f, 0.f};
  v8f acc1 = acc0, acc2 = acc0, acc3 = acc0;
  for (int kb = 0; kb < NQLR; kb += 32) {
    // A (16x32 bf16): lane<16 -> K0..15; lane>=16 -> K16..31 (contiguous)
    const __bf16* ap = arow + kb + half * 16;
    v8bf alo = *(const v8bf*)(ap);
    v8bf ahi = *(const v8bf*)(ap + 8);
    v16bf av = __builtin_shufflevector(alo, ahi, 0, 1, 2, 3, 4, 5, 6, 7,
                                       8, 9, 10, 11, 12, 13, 14, 15);
    // B (32x16 bf16): lane<16 -> K0..15; lane>=16 -> K16..31 (contiguous)
    v16bf b0 = *(const v16bf*)(brow[0] + kb);
    v16bf b1 = *(const v16bf*)(brow[1] + kb);
    v16bf b2 = *(const v16bf*)(brow[2] + kb);
    v16bf b3 = *(const v16bf*)(brow[3] + kb);
    acc0 = __builtin_amdgcn_wmma_f32_16x16x32_bf16(false, av, false, b0,
                                                   (short)0, acc0, false, false);
    acc1 = __builtin_amdgcn_wmma_f32_16x16x32_bf16(false, av, false, b1,
                                                   (short)0, acc1, false, false);
    acc2 = __builtin_amdgcn_wmma_f32_16x16x32_bf16(false, av, false, b2,
                                                   (short)0, acc2, false, false);
    acc3 = __builtin_amdgcn_wmma_f32_16x16x32_bf16(false, av, false, b3,
                                                   (short)0, acc3, false, false);
  }
  // D layout: vgpr r -> M = r + 8*half, N = l16
  float* outb = qraw + (size_t)(m_base + half * 8) * NHHD + n_base + l16;
#pragma unroll
  for (int r = 0; r < 8; ++r) {
    outb[(size_t)r * NHHD + 0]  = acc0[r];
    outb[(size_t)r * NHHD + 16] = acc1[r];
    outb[(size_t)r * NHHD + 32] = acc2[r];
    outb[(size_t)r * NHHD + 48] = acc3[r];
  }
}

// ============================================================
// Kernel 3: RoPE on q (first 64 of each head, pairs 32 apart) -> bf16
// ============================================================
__global__ void __launch_bounds__(256) rope_q(
    const float* __restrict__ qraw, const float* __restrict__ fcos,
    const float* __restrict__ fsin, __bf16* __restrict__ qb) {
  int i = blockIdx.x * 256 + threadIdx.x;   // exact: B*S*H*HD threads
  int tok = i >> 13;                        // 8192 elems per token
  int s = tok & (NS - 1);
  int hd = i & (NHD - 1);
  float v;
  if (hd < 32) {
    float c = fcos[s * 32 + hd], sn = fsin[s * 32 + hd];
    v = qraw[i] * c - qraw[i + 32] * sn;
  } else if (hd < 64) {
    int j = hd - 32;
    float c = fcos[s * 32 + j], sn = fsin[s * 32 + j];
    v = qraw[i - 32] * sn + qraw[i] * c;
  } else {
    v = qraw[i];
  }
  qb[i] = (__bf16)v;
}

// ============================================================
// Kernel 4: index_score[b,s,t] = sum_h w[b,s,h]*relu(q[b,s,h,:]·k[b,t,:]) + mask
// Block = (b, s_tile, 8 t_tiles). q chunk (8 heads x 16 s x 128 d) staged
// into LDS via GLOBAL_LOAD_ASYNC_TO_LDS_B128; k tile in registers;
// 256 WMMAs per 16x16 output tile with relu/weight epilogue.
// ============================================================
__global__ void __launch_bounds__(256) score_kernel(
    const __bf16* __restrict__ qb, const __bf16* __restrict__ kb,
    const float* __restrict__ wgt, float* __restrict__ score) {
  __shared__ __bf16 qs[8 * 16 * NHD];   // 32 KB
  __shared__ float wls[16 * NH];        // 4 KB
  const int tid = threadIdx.x;
  const int lane = tid & 31, wave = tid >> 5;
  const int half = lane >> 4, l16 = lane & 15;
  const int b = blockIdx.z;
  const int s0 = blockIdx.y * 16;
  const int t0 = (blockIdx.x * 8 + wave) * 16;

  for (int i = tid; i < 16 * NH; i += 256)
    wls[i] = wgt[(size_t)(b * NS + s0 + (i >> 6)) * NH + (i & 63)];

  // k tile: B operand, 4 K-chunks of 32; lane<16 holds K base+0..15, lane>=16 base+16..31
  v16bf kreg[4];
  const __bf16* krow = kb + (size_t)(b * NS + t0 + l16) * NHD + half * 16;
#pragma unroll
  for (int c = 0; c < 4; ++c) kreg[c] = *(const v16bf*)(krow + c * 32);

  float sc[8] = {0, 0, 0, 0, 0, 0, 0, 0};

  for (int hc = 0; hc < 8; ++hc) {
    __syncthreads();
    // stage q chunk: heads [hc*8, hc*8+8), 16 s rows, 128 d (2048 x 16B)
    for (int i = tid; i < 2048; i += 256) {
      int row = i >> 4;          // 0..127 : (h_loc, s_loc)
      int h_loc = row >> 4;
      int s_loc = row & 15;
      int d8 = i & 15;
      const __bf16* src = qb +
          ((size_t)(b * NS + s0 + s_loc) * NH + hc * 8 + h_loc) * NHD + d8 * 8;
      __bf16* dst = qs + row * NHD + d8 * 8;
#if USE_ASYNC_LDS
      __builtin_amdgcn_global_load_async_to_lds_b128(
          (as1_b128_t)(uintptr_t)(const void*)src,
          (as3_b128_t)(uint32_t)(uintptr_t)(void*)dst,
          0, 0);
#else
      *(v8bf*)dst = *(const v8bf*)src;
#endif
    }
#if USE_ASYNC_LDS
    __builtin_amdgcn_s_wait_asynccnt(0);
#endif
    __syncthreads();
#pragma unroll
    for (int hl = 0; hl < 8; ++hl) {
      v8f acc = {0.f, 0.f, 0.f, 0.f, 0.f, 0.f, 0.f, 0.f};
      const __bf16* abase = qs + (hl * 16 + l16) * NHD + half * 16;
#pragma unroll
      for (int c = 0; c < 4; ++c) {
        v8bf lo = *(const v8bf*)(abase + c * 32);
        v8bf hi = *(const v8bf*)(abase + c * 32 + 8);
        v16bf av = __builtin_shufflevector(lo, hi, 0, 1, 2, 3, 4, 5, 6, 7,
                                           8, 9, 10, 11, 12, 13, 14, 15);
        acc = __builtin_amdgcn_wmma_f32_16x16x32_bf16(false, av, false, kreg[c],
                                                      (short)0, acc, false, false);
      }
      const int h = hc * 8 + hl;
#pragma unroll
      for (int r = 0; r < 8; ++r) {
        float l = acc[r];
        l = l > 0.f ? l : 0.f;                       // relu
        sc[r] += l * wls[(r + half * 8) * NH + h];   // weight by w[b,s,h]
      }
    }
  }
#pragma unroll
  for (int r = 0; r < 8; ++r) {
    int sg = s0 + r + half * 8;
    int tg = t0 + l16;
    float v = sc[r] + (tg <= sg ? 0.f : kNegInf);    // causal mask add
    score[(size_t)(b * NS + sg) * NS + tg] = v;
  }
}

// ============================================================
// Kernel 5: per-row top-512 of 1024 via bitonic sort in LDS.
// Comparator matches lax.top_k: descending value, ties -> lower index first.
// ============================================================
__global__ void __launch_bounds__(256) topk_kernel(
    const float* __restrict__ score, int* __restrict__ idx_out) {
  __shared__ float sv[NS];
  __shared__ int si[NS];
  const int row = blockIdx.x;  // b*S + s
  const int tid = threadIdx.x;
  for (int i = tid; i < NS; i += 256) {
    sv[i] = score[(size_t)row * NS + i];
    si[i] = i;
  }
  for (int k = 2; k <= NS; k <<= 1) {
    for (int j = k >> 1; j > 0; j >>= 1) {
      __syncthreads();
      for (int i = tid; i < NS; i += 256) {
        int p = i ^ j;
        if (p > i) {
          float va = sv[i], vb = sv[p];
          int ia = si[i], ib = si[p];
          bool before = (va > vb) || (va == vb && ia < ib);
          bool desc = ((i & k) == 0);
          if (desc ? !before : before) {
            sv[i] = vb; sv[p] = va;
            si[i] = ib; si[p] = ia;
          }
        }
      }
    }
  }
  __syncthreads();
  for (int i = tid; i < NTOPK; i += 256)
    idx_out[(size_t)row * NTOPK + i] = si[i];
}

// ============================================================
extern "C" void kernel_launch(void* const* d_in, const int* in_sizes, int n_in,
                              void* d_out, int out_size, void* d_ws, size_t ws_size,
                              hipStream_t stream) {
  const float* x     = (const float*)d_in[0];
  const float* qr    = (const float*)d_in[1];
  const float* fcos  = (const float*)d_in[2];
  const float* fsin  = (const float*)d_in[3];
  // d_in[4] = mask (causal) — synthesized in score_kernel
  const float* wq    = (const float*)d_in[5];
  const float* wk    = (const float*)d_in[6];
  const float* lnw   = (const float*)d_in[7];
  const float* lnb   = (const float*)d_in[8];
  const float* wproj = (const float*)d_in[9];
  // d_in[10] = index_topk (== 512, hardcoded)

  char* ws = (char*)d_ws;
  size_t off = 0;
  float*  qraw = (float*)(ws + off);  off += (size_t)NB * NS * NHHD * 4;   // 64 MB
  __bf16* qb   = (__bf16*)(ws + off); off += (size_t)NB * NS * NHHD * 2;   // 32 MB
  __bf16* kbuf = (__bf16*)(ws + off); off += (size_t)NB * NS * NHD * 2;    // 512 KB
  float*  wgt  = (float*)(ws + off);  off += (size_t)NB * NS * NH * 4;     // 512 KB
  __bf16* qrb  = (__bf16*)(ws + off); off += (size_t)NB * NS * NQLR * 2;   // 6 MB
  __bf16* wqb  = (__bf16*)(ws + off); off += (size_t)NHHD * NQLR * 2;      // 24 MB

  int*   idx_out   = (int*)d_out;
  float* score_out = (float*)d_out + (size_t)NB * NS * NTOPK;

  // operand conversion (qr: 3.1M elems, wq: 12.6M elems; grids exact)
  f32_to_bf16<<<(NB * NS * NQLR) / 256, 256, 0, stream>>>(qr, qrb);
  f32_to_bf16<<<(NHHD * NQLR) / 256, 256, 0, stream>>>(wq, wqb);

  prep_kw<<<NB * NS, 256, 0, stream>>>(x, wk, lnw, lnb, wproj, fcos, fsin,
                                       kbuf, wgt);
  qgemm<<<dim3(NHHD / 512, (NB * NS) / 16), 256, 0, stream>>>(qrb, wqb, qraw);
  rope_q<<<(NB * NS * NHHD) / 256, 256, 0, stream>>>(qraw, fcos, fsin, qb);
  score_kernel<<<dim3(NS / 128, NS / 16, NB), 256, 0, stream>>>(qb, kbuf, wgt,
                                                                score_out);
  topk_kernel<<<NB * NS, 256, 0, stream>>>(score_out, idx_out);
}
